// GraphNetwork_36077725286786
// MI455X (gfx1250) — compile-verified, matrix operand order checked by reading, not run on hardware
//
#include <hip/hip_runtime.h>
#include <math.h>

#define NN 50000
#define INDIM 128
#define HIDD 128
#define HEADS 8
#define CD 16
#define NLAYERS 3
#define EE 800000
#define GG 8
#define NEG_SLOPE 0.2f
#define LN_EPS 1e-5f
#define AS_LD 136   // padded LDS row stride in halves (68 dwords -> conflict-free)

typedef __attribute__((ext_vector_type(16))) _Float16 v16h;
typedef __attribute__((ext_vector_type(8)))  _Float16 v8h;
typedef __attribute__((ext_vector_type(8)))  float    v8f;

// ---------------- weight pre-conversion: WT[n*128+k] = (f16)W[k*128+n] ----------------
__global__ void k_convW(const float* __restrict__ W, _Float16* __restrict__ WT) {
    int idx = blockIdx.x * blockDim.x + threadIdx.x;   // 16384 = 128*128
    int n = idx >> 7, k = idx & 127;
    WT[idx] = (_Float16)W[k * HIDD + n];
}

// ---------------- WMMA GEMM core ----------------
// A fragment from LDS (16x32 f16 per chunk): lane<16 row M=lane, K=kb+0..7, kb+16..23
// B fragment from f16 transposed weights: lane owns output column n, same K striping.
__device__ __forceinline__ v8f gemm_tile_lds(const _Float16* As, const _Float16* __restrict__ WT,
                                             int col0, int lane) {
    int mm = lane & 15;
    int n  = col0 + mm;
    int kshift = (lane & 16) ? 8 : 0;
    v8f acc = {};
#pragma unroll
    for (int kc = 0; kc < 4; ++kc) {
        int kb = kc * 32 + kshift;
        union { v16h v; v8h h[2]; } a, b;
        a.h[0] = *(const v8h*)(As + mm * AS_LD + kb);
        a.h[1] = *(const v8h*)(As + mm * AS_LD + kb + 16);
        b.h[0] = *(const v8h*)(WT + (size_t)n * HIDD + kb);
        b.h[1] = *(const v8h*)(WT + (size_t)n * HIDD + kb + 16);
        acc = __builtin_amdgcn_wmma_f32_16x16x32_f16(false, a.v, false, b.v,
                                                     (short)0, acc, false, false);
    }
    return acc;
}

// Cooperative f32->f16 staging of a 16x128 activation tile into LDS.
__device__ __forceinline__ void stage_A(const float* __restrict__ X, int row0, _Float16* As) {
    int t = threadIdx.x;               // 256 threads, each converts 8 contiguous floats
    int m = t >> 4, seg = t & 15;
    const float* px = X + (size_t)(row0 + m) * HIDD + seg * 8;
    v8h tmp;
#pragma unroll
    for (int e = 0; e < 8; ++e) tmp[e] = (_Float16)px[e];
    *(v8h*)(As + m * AS_LD + seg * 8) = tmp;
    __syncthreads();
}

// ---------------- GEMM kernels (block = 256 thr = 8 waves, grid = N/16) ----------------
__global__ void k_embed(const float* __restrict__ x, const _Float16* __restrict__ WT,
                        const float* __restrict__ bias, const float* __restrict__ type_emb,
                        const int* __restrict__ types, float* __restrict__ out) {
    __shared__ __align__(16) _Float16 As[16 * AS_LD];
    int lane = threadIdx.x & 31, wave = threadIdx.x >> 5;
    int row0 = blockIdx.x * 16, col0 = wave * 16;
    stage_A(x, row0, As);
    v8f acc = gemm_tile_lds(As, WT, col0, lane);
    int n = col0 + (lane & 15);
    int mOff = (lane & 16) ? 8 : 0;
#pragma unroll
    for (int r = 0; r < 8; ++r) {
        int row = row0 + mOff + r;
        out[(size_t)row * HIDD + n] = acc[r] + bias[n] + type_emb[types[row] * HIDD + n];
    }
}

__global__ void k_proj(const float* __restrict__ H, const _Float16* __restrict__ WT,
                       float* __restrict__ out) {
    __shared__ __align__(16) _Float16 As[16 * AS_LD];
    int lane = threadIdx.x & 31, wave = threadIdx.x >> 5;
    int row0 = blockIdx.x * 16, col0 = wave * 16;
    stage_A(H, row0, As);
    v8f acc = gemm_tile_lds(As, WT, col0, lane);
    int n = col0 + (lane & 15);
    int mOff = (lane & 16) ? 8 : 0;
#pragma unroll
    for (int r = 0; r < 8; ++r)
        out[(size_t)(row0 + mOff + r) * HIDD + n] = acc[r];
}

__global__ void k_ga1(const float* __restrict__ H, const _Float16* __restrict__ WT,
                      const float* __restrict__ b1, float* __restrict__ out) {
    __shared__ __align__(16) _Float16 As[16 * AS_LD];
    int lane = threadIdx.x & 31, wave = threadIdx.x >> 5;
    int row0 = blockIdx.x * 16, col0 = wave * 16;
    stage_A(H, row0, As);
    v8f acc = gemm_tile_lds(As, WT, col0, lane);
    int n = col0 + (lane & 15);
    int mOff = (lane & 16) ? 8 : 0;
#pragma unroll
    for (int r = 0; r < 8; ++r)
        out[(size_t)(row0 + mOff + r) * HIDD + n] = tanhf(acc[r] + b1[n]);
}

// ---------------- small helpers ----------------
__global__ void k_fill(float* __restrict__ p, long n, float v) {
    long i = (long)blockIdx.x * blockDim.x + threadIdx.x;
    if (i < n) p[i] = v;
}

__device__ __forceinline__ void atomicMaxF(float* addr, float v) {
    if (v >= 0.0f) atomicMax((int*)addr, __float_as_int(v));
    else           atomicMin((unsigned int*)addr, __float_as_uint(v));
}

__device__ __forceinline__ void edge_sd(const int* __restrict__ ei, int e, int& s, int& d) {
    if (e < EE) { s = ei[e]; d = ei[EE + e]; }
    else        { s = e - EE; d = s; }     // self loop
}

// per-node attention dot products sa/da [N,HEADS]
__global__ void k_attdot(const float* __restrict__ xp, const float* __restrict__ att_src,
                         const float* __restrict__ att_dst, float* __restrict__ sa,
                         float* __restrict__ da) {
    int idx = blockIdx.x * blockDim.x + threadIdx.x;
    if (idx >= NN * HEADS) return;
    int n = idx >> 3, hh = idx & 7;
    const float* p  = xp + (size_t)n * HIDD + hh * CD;
    const float* as = att_src + hh * CD;
    const float* ad = att_dst + hh * CD;
    float s = 0.0f, d = 0.0f;
#pragma unroll
    for (int c = 0; c < CD; ++c) { s += p[c] * as[c]; d += p[c] * ad[c]; }
    sa[idx] = s; da[idx] = d;
}

__device__ __forceinline__ float edge_alpha(const float* sa, const float* da, int s, int d, int hh) {
    float a = sa[s * HEADS + hh] + da[d * HEADS + hh];
    return a > 0.0f ? a : NEG_SLOPE * a;
}

__global__ void k_edge_max(const int* __restrict__ ei, const float* __restrict__ sa,
                           const float* __restrict__ da, float* __restrict__ amax) {
    int idx = blockIdx.x * blockDim.x + threadIdx.x;
    if (idx >= (EE + NN) * HEADS) return;
    int e = idx >> 3, hh = idx & 7, s, d;
    edge_sd(ei, e, s, d);
    atomicMaxF(&amax[d * HEADS + hh], edge_alpha(sa, da, s, d, hh));
}

__global__ void k_edge_sum(const int* __restrict__ ei, const float* __restrict__ sa,
                           const float* __restrict__ da, const float* __restrict__ amax,
                           float* __restrict__ denom) {
    int idx = blockIdx.x * blockDim.x + threadIdx.x;
    if (idx >= (EE + NN) * HEADS) return;
    int e = idx >> 3, hh = idx & 7, s, d;
    edge_sd(ei, e, s, d);
    float a = edge_alpha(sa, da, s, d, hh);
    atomicAdd(&denom[d * HEADS + hh], __expf(a - amax[d * HEADS + hh]));
}

__global__ void k_edge_agg(const int* __restrict__ ei, const float* __restrict__ sa,
                           const float* __restrict__ da, const float* __restrict__ amax,
                           const float* __restrict__ denom, const float* __restrict__ xp,
                           float* __restrict__ O) {
    int idx = blockIdx.x * blockDim.x + threadIdx.x;
    if (idx >= (EE + NN) * HEADS) return;
    int e = idx >> 3, hh = idx & 7, s, d;
    edge_sd(ei, e, s, d);
    float a = edge_alpha(sa, da, s, d, hh);
    float attn = __expf(a - amax[d * HEADS + hh]) / (denom[d * HEADS + hh] + 1e-16f);
    const float4* ps = (const float4*)(xp + (size_t)s * HIDD + hh * CD);
    float* po = O + (size_t)d * HIDD + hh * CD;
#pragma unroll
    for (int q = 0; q < 4; ++q) {
        float4 v = ps[q];
        atomicAdd(po + q * 4 + 0, v.x * attn);
        atomicAdd(po + q * 4 + 1, v.y * attn);
        atomicAdd(po + q * 4 + 2, v.z * attn);
        atomicAdd(po + q * 4 + 3, v.w * attn);
    }
}

// bias + LayerNorm + residual + ReLU, one wave per node (block = 128 thr = 4 nodes)
__global__ void k_ln(const float* __restrict__ O, const float* __restrict__ gat_b,
                     const float* __restrict__ g, const float* __restrict__ b,
                     float* __restrict__ H) {
    int wave = threadIdx.x >> 5, lane = threadIdx.x & 31;
    int n = blockIdx.x * 4 + wave;
    if (n >= NN) return;
    const float* po = O + (size_t)n * HIDD;
    float* ph = H + (size_t)n * HIDD;
    float v[4], s = 0.0f;
#pragma unroll
    for (int q = 0; q < 4; ++q) { int c = lane * 4 + q; v[q] = po[c] + gat_b[c]; s += v[q]; }
#pragma unroll
    for (int off = 16; off > 0; off >>= 1) s += __shfl_xor(s, off, 32);
    float mean = s * (1.0f / HIDD);
    float vs = 0.0f;
#pragma unroll
    for (int q = 0; q < 4; ++q) { float dd = v[q] - mean; vs += dd * dd; }
#pragma unroll
    for (int off = 16; off > 0; off >>= 1) vs += __shfl_xor(vs, off, 32);
    float inv = rsqrtf(vs * (1.0f / HIDD) + LN_EPS);
#pragma unroll
    for (int q = 0; q < 4; ++q) {
        int c = lane * 4 + q;
        float val = (v[q] - mean) * inv * g[c] + b[c] + ph[c];
        ph[c] = val > 0.0f ? val : 0.0f;
    }
}

// per-node logit = t . W2 + b2 ; global max (one wave per node)
__global__ void k_logit(const float* __restrict__ T, const float* __restrict__ W2,
                        const float* __restrict__ b2, float* __restrict__ logit,
                        float* __restrict__ gmax) {
    int wave = threadIdx.x >> 5, lane = threadIdx.x & 31;
    int n = blockIdx.x * 4 + wave;
    if (n >= NN) return;
    const float* p = T + (size_t)n * HIDD;
    float s = 0.0f;
#pragma unroll
    for (int q = 0; q < 4; ++q) { int c = lane * 4 + q; s += p[c] * W2[c]; }
#pragma unroll
    for (int off = 16; off > 0; off >>= 1) s += __shfl_xor(s, off, 32);
    if (lane == 0) {
        float l = s + b2[0];
        logit[n] = l;
        atomicMaxF(gmax, l);
    }
}

__global__ void k_expsum(const float* __restrict__ gmax, float* __restrict__ logit,
                         float* __restrict__ gsum, const int* __restrict__ batch,
                         float* __restrict__ cnt) {
    int n = blockIdx.x * blockDim.x + threadIdx.x;
    if (n >= NN) return;
    float ex = __expf(logit[n] - gmax[0]);
    logit[n] = ex;
    atomicAdd(gsum, ex);
    atomicAdd(&cnt[batch[n]], 1.0f);
}

// weighted global sum + per-graph mean pool (block = 128 chans, 64 nodes per block)
__global__ void k_pool(const float* __restrict__ H, const float* __restrict__ ex,
                       const float* __restrict__ gsum, const int* __restrict__ batch,
                       float* __restrict__ hg, float* __restrict__ pool) {
    int j = threadIdx.x;
    int base = blockIdx.x * 64;
    float inv = 1.0f / gsum[0];
    float acc = 0.0f;
    for (int i = 0; i < 64; ++i) {
        int n = base + i;
        if (n >= NN) break;
        float hv = H[(size_t)n * HIDD + j];
        acc += hv * ex[n] * inv;
        atomicAdd(&pool[batch[n] * HIDD + j], hv);
    }
    atomicAdd(&hg[j], acc);
}

__global__ void k_final(const float* __restrict__ hg, const float* __restrict__ pool,
                        const float* __restrict__ cnt, float* __restrict__ out) {
    int idx = blockIdx.x * blockDim.x + threadIdx.x;
    if (idx >= GG * HIDD) return;
    int g = idx / HIDD, j = idx % HIDD;
    float c = cnt[g]; c = c > 1.0f ? c : 1.0f;
    out[idx] = hg[j] + pool[idx] / c;
}

// ---------------- orchestration ----------------
extern "C" void kernel_launch(void* const* d_in, const int* in_sizes, int n_in,
                              void* d_out, int out_size, void* d_ws, size_t ws_size,
                              hipStream_t stream) {
    const float* x        = (const float*)d_in[0];
    const int*   ei       = (const int*)  d_in[1];
    const int*   batch    = (const int*)  d_in[2];
    const int*   types    = (const int*)  d_in[3];
    const float* emb_W    = (const float*)d_in[4];
    const float* emb_b    = (const float*)d_in[5];
    const float* type_emb = (const float*)d_in[6];
    const float* gat_W    = (const float*)d_in[7];
    const float* att_src  = (const float*)d_in[8];
    const float* att_dst  = (const float*)d_in[9];
    const float* gat_b    = (const float*)d_in[10];
    const float* ln_g     = (const float*)d_in[11];
    const float* ln_b     = (const float*)d_in[12];
    const float* ga_W1    = (const float*)d_in[13];
    const float* ga_b1    = (const float*)d_in[14];
    const float* ga_W2    = (const float*)d_in[15];
    const float* ga_b2    = (const float*)d_in[16];

    float* ws = (float*)d_ws;
    size_t off = 0;
    auto alloc = [&](size_t n) { float* p = ws + off; off += n; return p; };
    float* H     = alloc((size_t)NN * HIDD);
    float* XP    = alloc((size_t)NN * HIDD);
    float* O     = alloc((size_t)NN * HIDD);
    float* sa    = alloc((size_t)NN * HEADS);
    float* da    = alloc((size_t)NN * HEADS);
    float* amax  = alloc((size_t)NN * HEADS);
    float* denom = alloc((size_t)NN * HEADS);
    float* logit = alloc((size_t)NN);
    float* hg    = alloc(HIDD);        // hg..gsum contiguous (zero-filled together)
    float* pool  = alloc((size_t)GG * HIDD);
    float* cnt   = alloc(GG);
    float* gmax  = alloc(1);
    float* gsum  = alloc(1);
    off = (off + 3) & ~(size_t)3;      // 16-byte align for f16 weight buffers
    _Float16* wtEmb = (_Float16*)(ws + off);           // 5 x 128x128 f16 (transposed)
    _Float16* wtGat = wtEmb + 128 * 128;               // 3 layers
    _Float16* wtGa1 = wtGat + 3 * 128 * 128;

    const long NH  = (long)NN * HIDD;
    const int  ETH = (EE + NN) * HEADS;        // 6.8M edge-head work items
    dim3 gGrid(NN / 16), gBlock(256);          // N = 3125*16 exactly

    // one-time (per launch) f32 -> f16 transposed weight conversion
    k_convW<<<64, 256, 0, stream>>>(emb_W, wtEmb);
    for (int l = 0; l < NLAYERS; ++l)
        k_convW<<<64, 256, 0, stream>>>(gat_W + (size_t)l * HIDD * HIDD,
                                        wtGat + (size_t)l * HIDD * HIDD);
    k_convW<<<64, 256, 0, stream>>>(ga_W1, wtGa1);

    // node embedding (WMMA)
    k_embed<<<gGrid, gBlock, 0, stream>>>(x, wtEmb, emb_b, type_emb, types, H);

    for (int l = 0; l < NLAYERS; ++l) {
        k_proj<<<gGrid, gBlock, 0, stream>>>(H, wtGat + (size_t)l * HIDD * HIDD, XP);
        k_attdot<<<(NN * HEADS + 255) / 256, 256, 0, stream>>>(
            XP, att_src + l * HEADS * CD, att_dst + l * HEADS * CD, sa, da);
        k_fill<<<(int)((NH + 255) / 256), 256, 0, stream>>>(O, NH, 0.0f);
        k_fill<<<(NN * HEADS + 255) / 256, 256, 0, stream>>>(amax, (long)NN * HEADS, -INFINITY);
        k_fill<<<(NN * HEADS + 255) / 256, 256, 0, stream>>>(denom, (long)NN * HEADS, 0.0f);
        k_edge_max<<<(ETH + 255) / 256, 256, 0, stream>>>(ei, sa, da, amax);
        k_edge_sum<<<(ETH + 255) / 256, 256, 0, stream>>>(ei, sa, da, amax, denom);
        k_edge_agg<<<(ETH + 255) / 256, 256, 0, stream>>>(ei, sa, da, amax, denom, XP, O);
        k_ln<<<(NN + 3) / 4, 128, 0, stream>>>(O, gat_b + l * HIDD, ln_g + l * HIDD,
                                               ln_b + l * HIDD, H);
    }

    // global attention readout
    k_ga1<<<gGrid, gBlock, 0, stream>>>(H, wtGa1, ga_b1, XP);
    // zero hg, pool, cnt, gmax, gsum (contiguous: 128 + 1024 + 8 + 1 + 1 = 1162)
    k_fill<<<(1162 + 255) / 256, 256, 0, stream>>>(hg, 1162, 0.0f);
    k_fill<<<1, 32, 0, stream>>>(gmax, 1, -INFINITY);
    k_logit<<<(NN + 3) / 4, 128, 0, stream>>>(XP, ga_W2, ga_b2, logit, gmax);
    k_expsum<<<(NN + 255) / 256, 256, 0, stream>>>(gmax, logit, gsum, batch, cnt);
    k_pool<<<(NN + 63) / 64, 128, 0, stream>>>(H, logit, gsum, batch, hg, pool);
    k_final<<<(GG * HIDD + 255) / 256, 256, 0, stream>>>(hg, pool, cnt, (float*)d_out);
}
